// TemporalAttentionBlock_4183298146577
// MI455X (gfx1250) — matrix operand order repaired and checked by reading, not compiled, not run on hardware
//
#include <hip/hip_runtime.h>
#include <hip/hip_bf16.h>

typedef __attribute__((ext_vector_type(16))) _Float16 v16h;
typedef __attribute__((ext_vector_type(8)))  float    v8f;

// Problem constants (from reference)
constexpr int CB = 16, CT = 128, CV = 25, CD = 256, CH = 8, CMAXT = 300;
constexpr int HD = 32, DFF = 1024, NSEQ = CB * CV;
constexpr int BIASW = 2 * CMAXT - 1;          // 599
constexpr int LDX = CD + 8;                   // 264 halves, residual tile
constexpr int LDH = HD + 8;                   // 40 halves, per-head tiles
constexpr int LDP = CT + 8;                   // 136 halves, prob/GELU staging

// ---------------------------------------------------------------------------
// WMMA helpers (wave32): A and B both use contiguous k order, k = 16*(lane/16) + s
// ---------------------------------------------------------------------------
static __device__ __forceinline__ v8f wmma32(v16h a, v16h b, v8f c) {
  return __builtin_amdgcn_wmma_f32_16x16x32_f16(false, a, false, b, (short)0, c,
                                                false, false);
}

// A-fragment (16x32 f16), source row-major X[m][k], ld in halves.
static __device__ __forceinline__ v16h ldA(const _Float16* X, int ld, int m0,
                                           int k0, int lane) {
  const int hl = (lane >> 4) & 1;
  const _Float16* r = X + (m0 + (lane & 15)) * ld + k0 + hl * 16;
  v16h a;
#pragma unroll
  for (int p = 0; p < 4; ++p) { a[2 * p] = r[2 * p]; a[2 * p + 1] = r[2 * p + 1]; }
#pragma unroll
  for (int p = 0; p < 4; ++p) { a[8 + 2 * p] = r[8 + 2 * p]; a[9 + 2 * p] = r[8 + 2 * p + 1]; }
  return a;
}

// A-fragment with LayerNorm applied on the fly: xn = (x-mu)*rs*g + b
static __device__ __forceinline__ v16h lnA(const _Float16* X, int ld,
                                           const float* mean, const float* rstd,
                                           const float* g, const float* b,
                                           int m0, int k0, int lane) {
  const int hl = (lane >> 4) & 1;
  const int m = m0 + (lane & 15);
  const float mu = mean[m], rs = rstd[m];
  v16h a;
#pragma unroll
  for (int p = 0; p < 8; ++p) {
    const int kb = k0 + 2 * p + hl * 16;
#pragma unroll
    for (int e = 0; e < 2; ++e) {
      const int k = kb + e;
      const float xv = (float)X[m * ld + k];
      a[2 * p + e] = (_Float16)((xv - mu) * rs * g[k] + b[k]);
    }
  }
  return a;
}

// B-fragment (32x16) where B[k][n] = W[(n0+n)*ld + k0+k]  (W row-major [out][in])
static __device__ __forceinline__ v16h ldBT(const _Float16* W, int ld, int n0,
                                            int k0, int lane) {
  const int hl = (lane >> 4) & 1;
  const _Float16* r = W + (n0 + (lane & 15)) * ld + k0 + hl * 16;
  v16h b;
#pragma unroll
  for (int p = 0; p < 8; ++p) { b[2 * p] = r[2 * p]; b[2 * p + 1] = r[2 * p + 1]; }
  return b;
}

// B-fragment (32x16) where B[k][n] = X[(k0+k)*ld + n0+n]  (X row-major [k][n])
static __device__ __forceinline__ v16h ldBN(const _Float16* X, int ld, int k0,
                                            int n0, int lane) {
  const int hl = (lane >> 4) & 1;
  const _Float16* c = X + (k0 + hl * 16) * ld + n0 + (lane & 15);
  v16h b;
#pragma unroll
  for (int p = 0; p < 8; ++p) { b[2 * p] = c[(2 * p) * ld]; b[2 * p + 1] = c[(2 * p + 1) * ld]; }
  return b;
}

// ---------------------------------------------------------------------------
// fp32 -> fp16 weight conversion (weights are tiny; done once per launch)
// ---------------------------------------------------------------------------
__global__ void cvt_f32_to_f16(const float* __restrict__ src,
                               _Float16* __restrict__ dst, int n) {
  const int i = blockIdx.x * blockDim.x + threadIdx.x;
  if (i < n) dst[i] = (_Float16)src[i];
}

// ---------------------------------------------------------------------------
// Fused transformer block: one workgroup (8 wave32) per sequence n = b*V + v
// ---------------------------------------------------------------------------
__global__ __launch_bounds__(256) void tab_fused(
    const float* __restrict__ x, const float* __restrict__ ln1_g,
    const float* __restrict__ ln1_b, const float* __restrict__ b_qkv,
    const float* __restrict__ b_o, const float* __restrict__ ln2_g,
    const float* __restrict__ ln2_b, const float* __restrict__ b1,
    const float* __restrict__ b2, const float* __restrict__ relpos,
    const _Float16* __restrict__ wqkv16, const _Float16* __restrict__ wo16,
    const _Float16* __restrict__ w116, const _Float16* __restrict__ w216,
    float* __restrict__ out) {
  const int n = blockIdx.x;          // sequence id
  const int bb = n / CV;
  const int vv = n % CV;
  const int tid = threadIdx.x;
  const int lane = tid & 31;
  const int wave = tid >> 5;         // 0..7
  const int m0 = wave * 16;          // this wave's 16 rows
  const int hl = (lane >> 4) & 1;
  const int nn = lane & 15;

  __shared__ _Float16 XF[CT * LDX];     // residual xf (f16)         67.6 KB
  __shared__ _Float16 KH[CT * LDH];     // per-head K                10.2 KB
  __shared__ _Float16 VH[CT * LDH];     // per-head V                10.2 KB
  __shared__ _Float16 STG[CT * LDH];    // per-head Q / O staging    10.2 KB
  __shared__ _Float16 PP[CT * LDP];     // probs / GELU chunk        34.8 KB
  __shared__ float MEAN[CT];
  __shared__ float RSTD[CT];

  // ---- Phase 1: load x rows (strided), LN1 statistics, residual to LDS ----
  for (int i = 0; i < 16; ++i) {
    const int row = m0 + i;
    const float* xr = x + ((((size_t)bb * CT + row) * CV) + vv) * CD;
    const float4* x4 = reinterpret_cast<const float4*>(xr);
    const float4 u0 = x4[lane * 2];
    const float4 u1 = x4[lane * 2 + 1];
    float s = u0.x + u0.y + u0.z + u0.w + u1.x + u1.y + u1.z + u1.w;
    float sq = u0.x * u0.x + u0.y * u0.y + u0.z * u0.z + u0.w * u0.w +
               u1.x * u1.x + u1.y * u1.y + u1.z * u1.z + u1.w * u1.w;
#pragma unroll
    for (int off = 16; off >= 1; off >>= 1) {
      s += __shfl_xor(s, off, 32);
      sq += __shfl_xor(sq, off, 32);
    }
    const float mu = s * (1.0f / CD);
    const float rs = rsqrtf(sq * (1.0f / CD) - mu * mu + 1e-5f);
    _Float16* xfr = XF + row * LDX + lane * 8;
    xfr[0] = (_Float16)u0.x; xfr[1] = (_Float16)u0.y;
    xfr[2] = (_Float16)u0.z; xfr[3] = (_Float16)u0.w;
    xfr[4] = (_Float16)u1.x; xfr[5] = (_Float16)u1.y;
    xfr[6] = (_Float16)u1.z; xfr[7] = (_Float16)u1.w;
    if (lane == 0) { MEAN[row] = mu; RSTD[row] = rs; }
  }
  __syncthreads();

  // Hoisted LN1(x) A-fragments for this wave's rows (K = 256 in 8 blocks)
  v16h aX[8];
#pragma unroll
  for (int kb = 0; kb < 8; ++kb)
    aX[kb] = lnA(XF, LDX, MEAN, RSTD, ln1_g, ln1_b, m0, kb * 32, lane);

  const float kScale = 0.17677669529663687f;  // 1/sqrt(32)

  // ---- Phase 2: attention, one head at a time (Q/K/V recomputed per head).
  // Each head's slice of the output projection is folded straight into the
  // residual tile XF (own rows only), so no 128-VGPR accumulator stays live.
  for (int h = 0; h < CH; ++h) {
    const int hc = h * HD;
    // Q_h / K_h / V_h for this wave's 16 rows (GEMM 16x256x32 each)
#pragma unroll
    for (int nt = 0; nt < 2; ++nt) {
      const int cn = nt * 16 + nn;
      v8f cQ, cK, cV;
      const float bq = b_qkv[hc + cn];
      const float bk = b_qkv[CD + hc + cn];
      const float bv = b_qkv[2 * CD + hc + cn];
#pragma unroll
      for (int r = 0; r < 8; ++r) { cQ[r] = bq; cK[r] = bk; cV[r] = bv; }
#pragma unroll
      for (int kb = 0; kb < 8; ++kb) {
        cQ = wmma32(aX[kb], ldBT(wqkv16, CD, hc + nt * 16, kb * 32, lane), cQ);
        cK = wmma32(aX[kb], ldBT(wqkv16, CD, CD + hc + nt * 16, kb * 32, lane), cK);
        cV = wmma32(aX[kb], ldBT(wqkv16, CD, 2 * CD + hc + nt * 16, kb * 32, lane), cV);
      }
#pragma unroll
      for (int r = 0; r < 8; ++r) {
        const int row = m0 + r + hl * 8;
        STG[row * LDH + cn] = (_Float16)cQ[r];
        KH[row * LDH + cn] = (_Float16)cK[r];
        VH[row * LDH + cn] = (_Float16)cV[r];
      }
    }
    __syncthreads();  // K_h / V_h are read across all waves

    const v16h aQ = ldA(STG, LDH, m0, 0, lane);
    float sreg[8][8];
#pragma unroll
    for (int st = 0; st < 8; ++st) {
      const v16h bK = ldBT(KH, LDH, st * 16, 0, lane);  // B[k][n] = K[s][k]
      v8f c;
#pragma unroll
      for (int r = 0; r < 8; ++r) c[r] = 0.0f;
      c = wmma32(aQ, bK, c);
#pragma unroll
      for (int r = 0; r < 8; ++r) {
        const int row = m0 + r + hl * 8;
        const int cg = st * 16 + nn;
        sreg[st][r] =
            c[r] * kScale + relpos[h * BIASW + (row - cg + (CMAXT - 1))];
      }
    }
    // Softmax over 128 cols: rows live in one 16-lane half; xor masks <16 stay there
    float mx[8], sm[8];
#pragma unroll
    for (int r = 0; r < 8; ++r) {
      float mv = sreg[0][r];
#pragma unroll
      for (int st = 1; st < 8; ++st) mv = fmaxf(mv, sreg[st][r]);
      mx[r] = mv;
    }
#pragma unroll
    for (int off = 8; off >= 1; off >>= 1)
#pragma unroll
      for (int r = 0; r < 8; ++r) mx[r] = fmaxf(mx[r], __shfl_xor(mx[r], off, 32));
#pragma unroll
    for (int r = 0; r < 8; ++r) {
      float s = 0.0f;
#pragma unroll
      for (int st = 0; st < 8; ++st) {
        sreg[st][r] = __expf(sreg[st][r] - mx[r]);
        s += sreg[st][r];
      }
      sm[r] = s;
    }
#pragma unroll
    for (int off = 8; off >= 1; off >>= 1)
#pragma unroll
      for (int r = 0; r < 8; ++r) sm[r] += __shfl_xor(sm[r], off, 32);
#pragma unroll
    for (int r = 0; r < 8; ++r) sm[r] = 1.0f / sm[r];
    // C-layout -> A-layout via per-wave LDS slice
#pragma unroll
    for (int st = 0; st < 8; ++st)
#pragma unroll
      for (int r = 0; r < 8; ++r)
        PP[(m0 + r + hl * 8) * LDP + st * 16 + nn] =
            (_Float16)(sreg[st][r] * sm[r]);

    // O_h = P @ V_h  (16x128x32)
    v8f co0, co1;
#pragma unroll
    for (int r = 0; r < 8; ++r) { co0[r] = 0.0f; co1[r] = 0.0f; }
#pragma unroll
    for (int kt = 0; kt < 4; ++kt) {
      const v16h aP = ldA(PP, LDP, m0, kt * 32, lane);
      co0 = wmma32(aP, ldBN(VH, LDH, kt * 32, 0, lane), co0);
      co1 = wmma32(aP, ldBN(VH, LDH, kt * 32, 16, lane), co1);
    }
    // Stage O_h, re-load as A-fragment
#pragma unroll
    for (int r = 0; r < 8; ++r) {
      const int row = m0 + r + hl * 8;
      STG[row * LDH + nn] = (_Float16)co0[r];
      STG[row * LDH + 16 + nn] = (_Float16)co1[r];
    }
    const v16h aO = ldA(STG, LDH, m0, 0, lane);
    // Fold this head's o-projection slice into the residual tile (own rows):
    // XF += O_h @ Wo[:, hc:hc+32]^T   (+ b_o once, at h == 0)
#pragma unroll
    for (int nt = 0; nt < 16; ++nt) {
      const int col = nt * 16 + nn;
      const float bo = (h == 0) ? b_o[col] : 0.0f;
      v8f c;
#pragma unroll
      for (int r = 0; r < 8; ++r)
        c[r] = (float)XF[(m0 + r + hl * 8) * LDX + col] + bo;
      c = wmma32(aO, ldBT(wo16, CD, nt * 16, hc, lane), c);
#pragma unroll
      for (int r = 0; r < 8; ++r)
        XF[(m0 + r + hl * 8) * LDX + col] = (_Float16)c[r];
    }
    __syncthreads();  // protect K_h/V_h before next head overwrites
  }

  // ---- Phase 3: LN2 statistics (values normalized on the fly) ----
  for (int i = 0; i < 16; ++i) {
    const int row = m0 + i;
    float s = 0.0f, sq = 0.0f;
#pragma unroll
    for (int e = 0; e < 8; ++e) {
      const float xv = (float)XF[row * LDX + lane * 8 + e];
      s += xv; sq += xv * xv;
    }
#pragma unroll
    for (int off = 16; off >= 1; off >>= 1) {
      s += __shfl_xor(s, off, 32);
      sq += __shfl_xor(sq, off, 32);
    }
    const float mu = s * (1.0f / CD);
    const float rs = rsqrtf(sq * (1.0f / CD) - mu * mu + 1e-5f);
    if (lane == 0) { MEAN[row] = mu; RSTD[row] = rs; }
  }

  v16h aH[8];
#pragma unroll
  for (int kb = 0; kb < 8; ++kb)
    aH[kb] = lnA(XF, LDX, MEAN, RSTD, ln2_g, ln2_b, m0, kb * 32, lane);

  // ---- Phase 4: FFN, hidden dim in 8 chunks of 128, accumulate in regs ----
  v8f acc[16];
#pragma unroll
  for (int nt = 0; nt < 16; ++nt) {
    const int col = nt * 16 + nn;
    const float bv = b2[col];
#pragma unroll
    for (int r = 0; r < 8; ++r)
      acc[nt][r] = (float)XF[(m0 + r + hl * 8) * LDX + col] + bv;
  }
  for (int ch = 0; ch < 8; ++ch) {
    const int hb = ch * 128;
#pragma unroll
    for (int gt = 0; gt < 8; ++gt) {
      const float b1v = b1[hb + gt * 16 + nn];
      v8f c;
#pragma unroll
      for (int r = 0; r < 8; ++r) c[r] = b1v;
#pragma unroll
      for (int kb = 0; kb < 8; ++kb)
        c = wmma32(aH[kb], ldBT(w116, CD, hb + gt * 16, kb * 32, lane), c);
#pragma unroll
      for (int r = 0; r < 8; ++r) {
        const float xg = c[r];  // exact GELU: 0.5*x*(1+erf(x/sqrt(2)))
        PP[(m0 + r + hl * 8) * LDP + gt * 16 + nn] =
            (_Float16)(0.5f * xg * (1.0f + erff(xg * 0.70710678118654752f)));
      }
    }
    v16h aG[4];
#pragma unroll
    for (int kt = 0; kt < 4; ++kt) aG[kt] = ldA(PP, LDP, m0, kt * 32, lane);
#pragma unroll
    for (int nt = 0; nt < 16; ++nt)
#pragma unroll
      for (int kt = 0; kt < 4; ++kt)
        acc[nt] = wmma32(aG[kt], ldBT(w216, DFF, nt * 16, hb + kt * 32, lane),
                         acc[nt]);
    if (ch < 7) {  // pull next weight chunk toward L1 (global_prefetch_b8)
      __builtin_prefetch(w116 + (size_t)(hb + 128) * CD, 0, 1);
      __builtin_prefetch(w216 + hb + 128, 0, 1);
    }
  }

  // ---- Phase 5: out[b][t][v][d] = xf + ffn (already contains xf + b2) -----
#pragma unroll
  for (int nt = 0; nt < 16; ++nt) {
    const int col = nt * 16 + nn;
#pragma unroll
    for (int r = 0; r < 8; ++r) {
      const int row = m0 + r + hl * 8;
      out[((((size_t)bb * CT + row) * CV) + vv) * CD + col] = acc[nt][r];
    }
  }
}

// ---------------------------------------------------------------------------
extern "C" void kernel_launch(void* const* d_in, const int* in_sizes, int n_in,
                              void* d_out, int out_size, void* d_ws,
                              size_t ws_size, hipStream_t stream) {
  (void)in_sizes; (void)n_in; (void)out_size; (void)ws_size;
  const float* x      = (const float*)d_in[0];
  const float* ln1_g  = (const float*)d_in[1];
  const float* ln1_b  = (const float*)d_in[2];
  const float* w_qkv  = (const float*)d_in[3];
  const float* b_qkv  = (const float*)d_in[4];
  const float* w_o    = (const float*)d_in[5];
  const float* b_o    = (const float*)d_in[6];
  const float* ln2_g  = (const float*)d_in[7];
  const float* ln2_b  = (const float*)d_in[8];
  const float* w1     = (const float*)d_in[9];
  const float* b1     = (const float*)d_in[10];
  const float* w2     = (const float*)d_in[11];
  const float* b2     = (const float*)d_in[12];
  const float* relpos = (const float*)d_in[13];
  float* out = (float*)d_out;

  // fp16 weight copies in workspace
  _Float16* ws = (_Float16*)d_ws;
  const int n_qkv = 3 * CD * CD;   // 196608
  const int n_o   = CD * CD;       //  65536
  const int n_w1  = DFF * CD;      // 262144
  const int n_w2  = CD * DFF;      // 262144
  _Float16* wqkv16 = ws;
  _Float16* wo16   = wqkv16 + n_qkv;
  _Float16* w116   = wo16 + n_o;
  _Float16* w216   = w116 + n_w1;

  cvt_f32_to_f16<<<(n_qkv + 255) / 256, 256, 0, stream>>>(w_qkv, wqkv16, n_qkv);
  cvt_f32_to_f16<<<(n_o   + 255) / 256, 256, 0, stream>>>(w_o,   wo16,   n_o);
  cvt_f32_to_f16<<<(n_w1  + 255) / 256, 256, 0, stream>>>(w1,    w116,   n_w1);
  cvt_f32_to_f16<<<(n_w2  + 255) / 256, 256, 0, stream>>>(w2,    w216,   n_w2);

  tab_fused<<<NSEQ, 256, 0, stream>>>(x, ln1_g, ln1_b, b_qkv, b_o, ln2_g,
                                      ln2_b, b1, b2, relpos, wqkv16, wo16,
                                      w116, w216, out);
}
